// RPN_27212912787728
// MI455X (gfx1250) — matrix operand order repaired and checked by reading, not compiled, not run on hardware
//
#include <hip/hip_runtime.h>
#include <math.h>

typedef __bf16 bf16;
typedef __attribute__((ext_vector_type(16))) __bf16 bf16x16;
typedef __attribute__((ext_vector_type(8)))  __bf16 bf16x8;
typedef __attribute__((ext_vector_type(8)))  float  f32x8;

#define NB 2
#define CCH 256
#define NA 3
#define TA 159882          // total anchors per image = sum HW*3
#define MCAND 4507         // 1000*4 + 507 candidates per image
#define NEG_INF (-10000000000.0f)
#define SCALE_CLAMP 4.135166556742356f   // log(1000/16)
#define PAD_KEY (-3.0e38f)

__constant__ int d_cnt[5]  = {120000, 30000, 7500, 1875, 507};
__constant__ int d_loff[5] = {0, 120000, 150000, 157500, 159375};
__constant__ int d_cumk[5] = {0, 1000, 2000, 3000, 4000};

// ---------------- WMMA fragment helpers (layouts per CDNA5 ISA 7.12.2) ----
union FragU { bf16x16 v; bf16x8 h[2]; };

static __device__ inline bf16x16 frag_a(const bf16* row, int half) {
  // A 16x32 bf16: VGPR0-3 = K half*8..half*8+7 ; VGPR4-7 = K 16+half*8..+7
  FragU f;
  f.h[0] = *reinterpret_cast<const bf16x8*>(row + half * 8);
  f.h[1] = *reinterpret_cast<const bf16x8*>(row + 16 + half * 8);
  return f.v;
}
static __device__ inline bf16x16 frag_b(const bf16* colb, int half) {
  // B 32x16 bf16 stored N-major: lane n holds K = half*8..half*8+7, 16+half*8..+7
  FragU f;
  f.h[0] = *reinterpret_cast<const bf16x8*>(colb + half * 8);
  f.h[1] = *reinterpret_cast<const bf16x8*>(colb + 16 + half * 8);
  return f.v;
}
#define WMMA_BF16(a, b, c) \
  __builtin_amdgcn_wmma_f32_16x16x32_bf16(false, (a), false, (b), (short)0, (c), false, false)

// ---------------- weight packing ------------------------------------------
__global__ __launch_bounds__(256) void pack_w_kernel(const float* __restrict__ conv_w,
                                                     bf16* __restrict__ wb2) {
  int idx = blockIdx.x * 256 + threadIdx.x;          // [kb=72][n=256][kl=32]
  if (idx >= 72 * 256 * 32) return;
  int kl = idx & 31;
  int n  = (idx >> 5) & 255;
  int kb = idx >> 13;
  int k  = kb * 32 + kl;
  int tap = k >> 8;                                  // ky*3+kx
  int ci  = k & 255;
  wb2[idx] = (bf16)conv_w[((size_t)(n * 256 + ci)) * 9 + tap];
}

__global__ __launch_bounds__(256) void pack_head_kernel(const float* __restrict__ obj_w,
                                                        const float* __restrict__ delta_w,
                                                        bf16* __restrict__ wh) {
  int idx = blockIdx.x * 256 + threadIdx.x;          // [kb=8][n=16][kl=32]
  if (idx >= 8 * 16 * 32) return;
  int kl = idx & 31;
  int n  = (idx >> 5) & 15;
  int kb = idx >> 9;
  int k  = kb * 32 + kl;
  float v = 0.0f;
  if (n < 3)       v = obj_w[n * 256 + k];
  else if (n < 15) v = delta_w[(n - 3) * 256 + k];
  wh[idx] = (bf16)v;
}

// ---------------- NCHW f32 -> padded NHWC bf16 ----------------------------
__global__ __launch_bounds__(256) void pad_kernel(const float* __restrict__ f,
                                                  bf16* __restrict__ pad, int H, int W) {
  long idx = (long)blockIdx.x * 256 + threadIdx.x;
  int Hp = H + 2, Wp = W + 2;
  long total = (long)NB * Hp * Wp * CCH;
  if (idx >= total) return;
  int c = (int)(idx & 255);
  long rest = idx >> 8;
  int px = (int)(rest % Wp); rest /= Wp;
  int py = (int)(rest % Hp);
  int img = (int)(rest / Hp);
  int yy = py - 1, xx = px - 1;
  float v = 0.0f;
  if (yy >= 0 && yy < H && xx >= 0 && xx < W)
    v = f[(((size_t)img * CCH + c) * H + yy) * W + xx];
  pad[idx] = (bf16)v;
}

// ---------------- fused 3x3 conv (implicit GEMM) + ReLU + head GEMM -------
__global__ __launch_bounds__(256) void conv_head_kernel(
    const bf16* __restrict__ pad, const bf16* __restrict__ wb2, const bf16* __restrict__ wh,
    const float* __restrict__ conv_b, const float* __restrict__ obj_b,
    const float* __restrict__ delta_b, float* __restrict__ hout,
    int H, int W, int HW, int Mp) {
  __shared__ __align__(16) bf16 tls[16 * 264];       // relu'd tile, stride keeps 16B align
  int tid = threadIdx.x;
  int lane = tid & 31, wv = tid >> 5;
  int lm = lane & 15, half = lane >> 4;
  int m0 = blockIdx.x * 16;
  int img = blockIdx.y;
  int Wp = W + 2;
  int m = m0 + lm; if (m >= HW) m = HW - 1;          // clamp for partial tiles
  int y = m / W, x = m - y * W;
  const bf16* ibase = pad + (size_t)img * (H + 2) * Wp * CCH;

  f32x8 acc0 = {}; f32x8 acc1 = {};
  for (int tap = 0; tap < 9; ++tap) {
    int dy = tap / 3, dx = tap - dy * 3;
    const bf16* arow = ibase + ((size_t)((y + dy) * Wp + (x + dx))) * CCH;
    for (int cb = 0; cb < 8; ++cb) {                 // K = 9 taps * 8 blocks of 32
      int kb = tap * 8 + cb;
      bf16x16 af = frag_a(arow + cb * 32, half);
      const bf16* bcol = wb2 + ((size_t)kb * 256 + wv * 32 + lm) * 32;
      __builtin_prefetch(bcol + 256 * 32, 0, 1);
      bf16x16 b0 = frag_b(bcol, half);
      bf16x16 b1 = frag_b(bcol + 16 * 32, half);
      acc0 = WMMA_BF16(af, b0, acc0);
      acc1 = WMMA_BF16(af, b1, acc1);
    }
  }
  // bias + relu -> LDS bf16 tile [16 rows][256 cout]
  float bc0 = conv_b[wv * 32 + lm];
  float bc1 = conv_b[wv * 32 + 16 + lm];
#pragma unroll
  for (int j = 0; j < 8; ++j) {                      // C layout: row = j + half*8
    int row = j + half * 8;
    tls[row * 264 + wv * 32 + lm]      = (bf16)fmaxf(acc0[j] + bc0, 0.0f);
    tls[row * 264 + wv * 32 + 16 + lm] = (bf16)fmaxf(acc1[j] + bc1, 0.0f);
  }
  __syncthreads();
  if (wv == 0) {                                     // head GEMM 16x256 x 256x16
    f32x8 hacc = {};
    for (int kb2 = 0; kb2 < 8; ++kb2) {
      bf16x16 af  = frag_a(tls + lm * 264 + kb2 * 32, half);
      bf16x16 bfm = frag_b(wh + (size_t)(kb2 * 16 + lm) * 32, half);
      hacc = WMMA_BF16(af, bfm, hacc);
    }
    float hb = (lm < 3) ? obj_b[lm] : ((lm < 15) ? delta_b[lm - 3] : 0.0f);
#pragma unroll
    for (int j = 0; j < 8; ++j) {
      int row = j + half * 8;
      int m2 = m0 + row;
      if (m2 < HW) hout[((size_t)img * Mp + m2) * 16 + lm] = hacc[j] + hb;
    }
  }
}

// ---------------- anchor decode + clip ------------------------------------
__global__ __launch_bounds__(256) void decode_kernel(
    const float* __restrict__ hout, float* __restrict__ scores, float* __restrict__ boxes,
    int H, int W, int HW, int Mp, float stride, float asize, int loff) {
  int idx = blockIdx.x * 256 + threadIdx.x;
  int tot = NB * HW * NA;
  if (idx >= tot) return;
  int a = idx % NA;
  int m = (idx / NA) % HW;
  int img = idx / (NA * HW);
  int y = m / W, x = m - y * W;
  const float* h = hout + ((size_t)img * Mp + m) * 16;
  float obj = h[a];
  float dxv = h[3 + a * 4 + 0];
  float dyv = h[3 + a * 4 + 1];
  float dwv = fminf(h[3 + a * 4 + 2], SCALE_CLAMP);
  float dhv = fminf(h[3 + a * 4 + 3], SCALE_CLAMP);
  const float ratios[3] = {0.5f, 1.0f, 2.0f};
  float r = ratios[a];
  float aw = sqrtf(asize * asize / r);
  float ah = aw * r;
  float cx = x * stride, cy = y * stride;
  float pcx = dxv * aw + cx;
  float pcy = dyv * ah + cy;
  float pw = expf(dwv) * aw;
  float ph = expf(dhv) * ah;
  float x1 = fminf(fmaxf(pcx - 0.5f * pw, 0.0f), 800.0f);
  float y1 = fminf(fmaxf(pcy - 0.5f * ph, 0.0f), 800.0f);
  float x2 = fminf(fmaxf(pcx + 0.5f * pw, 0.0f), 800.0f);
  float y2 = fminf(fmaxf(pcy + 0.5f * ph, 0.0f), 800.0f);
  size_t gi = (size_t)img * TA + loff + (size_t)m * NA + a;
  scores[gi] = obj;                                  // raw logit (valid-mask applied later)
  boxes[gi * 4 + 0] = x1; boxes[gi * 4 + 1] = y1;
  boxes[gi * 4 + 2] = x2; boxes[gi * 4 + 3] = y2;
}

// ---------------- per (image,level) iterative top-k -----------------------
__global__ __launch_bounds__(256) void topk_kernel(
    float* __restrict__ scores, const float* __restrict__ boxes,
    float* __restrict__ cs, float* __restrict__ cb, int* __restrict__ cl) {
  int l = blockIdx.x, img = blockIdx.y;
  int cnt = d_cnt[l], loff = d_loff[l];
  int k = cnt < 1000 ? cnt : 1000;
  int q0 = d_cumk[l];
  float* sc = scores + (size_t)img * TA + loff;
  const float* bx = boxes + ((size_t)img * TA + loff) * 4;
  __shared__ float sv[256];
  __shared__ int   si[256];
  int tid = threadIdx.x;
  for (int it = 0; it < k; ++it) {
    float bv = PAD_KEY; int bi = cnt;
    for (int i = tid; i < cnt; i += 256) {
      float v = sc[i];
      if (v > bv) { bv = v; bi = i; }                // ascending scan -> lowest idx on tie
    }
    sv[tid] = bv; si[tid] = bi;
    __syncthreads();
    for (int s = 128; s > 0; s >>= 1) {
      if (tid < s) {
        if (sv[tid + s] > sv[tid] ||
            (sv[tid + s] == sv[tid] && si[tid + s] < si[tid])) {
          sv[tid] = sv[tid + s]; si[tid] = si[tid + s];
        }
      }
      __syncthreads();
    }
    if (tid == 0) {
      int b0 = si[0]; float s0 = sv[0];
      size_t q = (size_t)img * MCAND + q0 + it;
      if (b0 < cnt) {
        sc[b0] = PAD_KEY;                            // exclude from next iterations
        float bx1 = bx[(size_t)b0 * 4 + 0], by1 = bx[(size_t)b0 * 4 + 1];
        float bx2 = bx[(size_t)b0 * 4 + 2], by2 = bx[(size_t)b0 * 4 + 3];
        bool valid = (bx2 - bx1 > 0.0f) && (by2 - by1 > 0.0f);
        cs[q] = valid ? s0 : NEG_INF;
        cb[q * 4 + 0] = bx1; cb[q * 4 + 1] = by1;
        cb[q * 4 + 2] = bx2; cb[q * 4 + 3] = by2;
        cl[q] = l;
      } else {
        cs[q] = NEG_INF;
        cb[q * 4 + 0] = 0; cb[q * 4 + 1] = 0; cb[q * 4 + 2] = 0; cb[q * 4 + 3] = 0;
        cl[q] = l;
      }
    }
    __syncthreads();
  }
}

// ---------------- per-image bitonic sort (desc, tie -> lower idx) ---------
__global__ __launch_bounds__(1024) void sort_kernel(
    const float* __restrict__ cs, const float* __restrict__ cb, const int* __restrict__ cl,
    float* __restrict__ ss, float* __restrict__ sb, int* __restrict__ sl) {
  __shared__ float key[8192];
  __shared__ int   val[8192];
  int img = blockIdx.x, tid = threadIdx.x;
  for (int i = tid; i < 8192; i += 1024) {
    key[i] = (i < MCAND) ? cs[(size_t)img * MCAND + i] : -3.3e38f;
    val[i] = i;
  }
  __syncthreads();
  for (int kk = 2; kk <= 8192; kk <<= 1) {
    for (int j = kk >> 1; j > 0; j >>= 1) {
      for (int i = tid; i < 8192; i += 1024) {
        int ixj = i ^ j;
        if (ixj > i) {
          float ka = key[i], kb = key[ixj];
          int va = val[i], vb = val[ixj];
          bool aFirst = (ka > kb) || (ka == kb && va < vb);
          bool desc = ((i & kk) == 0);
          if (desc ? !aFirst : aFirst) {
            key[i] = kb; key[ixj] = ka;
            val[i] = vb; val[ixj] = va;
          }
        }
      }
      __syncthreads();
    }
  }
  for (int i = tid; i < MCAND; i += 1024) {
    int v = val[i];
    ss[(size_t)img * MCAND + i] = key[i];
    if (v < MCAND) {
      sl[(size_t)img * MCAND + i] = cl[(size_t)img * MCAND + v];
      for (int d = 0; d < 4; ++d)
        sb[((size_t)img * MCAND + i) * 4 + d] = cb[((size_t)img * MCAND + v) * 4 + d];
    }
  }
}

// ---------------- per-image greedy NMS + final emit -----------------------
__global__ __launch_bounds__(1024) void nms_kernel(
    const float* __restrict__ ss, const float* __restrict__ sb, const int* __restrict__ sl,
    float* __restrict__ out) {
  __shared__ int keep[MCAND];
  __shared__ int rankv[MCAND];
  __shared__ int sng;
  int img = blockIdx.x, tid = threadIdx.x;
  size_t base = (size_t)img * MCAND;
  for (int i = tid; i < MCAND; i += 1024) keep[i] = 1;
  __syncthreads();
  for (int i = 0; i < MCAND - 1; ++i) {
    if (keep[i]) {                                   // uniform (shared) branch
      float oi = (float)sl[base + i] * 4096.0f;
      float ax1 = sb[(base + i) * 4 + 0] + oi;
      float ay1 = sb[(base + i) * 4 + 1] + oi;
      float ax2 = sb[(base + i) * 4 + 2] + oi;
      float ay2 = sb[(base + i) * 4 + 3] + oi;
      float areaA = (ax2 - ax1) * (ay2 - ay1);
      for (int jj = i + 1 + tid; jj < MCAND; jj += 1024) {
        if (!keep[jj]) continue;
        float oj = (float)sl[base + jj] * 4096.0f;
        float bx1 = sb[(base + jj) * 4 + 0] + oj;
        float by1 = sb[(base + jj) * 4 + 1] + oj;
        float bx2 = sb[(base + jj) * 4 + 2] + oj;
        float by2 = sb[(base + jj) * 4 + 3] + oj;
        float areaB = (bx2 - bx1) * (by2 - by1);
        float ix1 = fmaxf(ax1, bx1), iy1 = fmaxf(ay1, by1);
        float ix2 = fminf(ax2, bx2), iy2 = fminf(ay2, by2);
        float w = fmaxf(ix2 - ix1, 0.0f), h = fmaxf(iy2 - iy1, 0.0f);
        float inter = w * h;
        float iou = inter / (areaA + areaB - inter + 1e-9f);
        if (iou > 0.7f) keep[jj] = 0;
      }
    }
    __syncthreads();
  }
  if (tid == 0) {                                    // top_k over masked scores:
    int rg = 0, rb = 0;                              // kept (score>NEG_INF) first by score,
    for (int i = 0; i < MCAND; ++i) {                // then NEG_INF entries by index
      bool good = keep[i] && (ss[base + i] > NEG_INF);
      rankv[i] = good ? rg++ : (MCAND + rb++);
    }
    sng = rg;
  }
  __syncthreads();
  for (int i = tid; i < MCAND; i += 1024) {
    int r = rankv[i];
    r = (r < MCAND) ? r : (sng + r - MCAND);
    if (r < 1000) {
      bool good = keep[i] && (ss[base + i] > NEG_INF);
      float msv = good ? ss[base + i] : NEG_INF;
      out[((size_t)img * 1000 + r) * 5 + 0] = sb[(base + i) * 4 + 0];
      out[((size_t)img * 1000 + r) * 5 + 1] = sb[(base + i) * 4 + 1];
      out[((size_t)img * 1000 + r) * 5 + 2] = sb[(base + i) * 4 + 2];
      out[((size_t)img * 1000 + r) * 5 + 3] = sb[(base + i) * 4 + 3];
      out[((size_t)img * 1000 + r) * 5 + 4] = msv;
    }
  }
}

// ---------------- host launch ---------------------------------------------
extern "C" void kernel_launch(void* const* d_in, const int* in_sizes, int n_in,
                              void* d_out, int out_size, void* d_ws, size_t ws_size,
                              hipStream_t stream) {
  (void)in_sizes; (void)n_in; (void)out_size; (void)ws_size;
  const float* f[5];
  for (int i = 0; i < 5; ++i) f[i] = (const float*)d_in[i];
  const float* conv_w  = (const float*)d_in[5];
  const float* conv_b  = (const float*)d_in[6];
  const float* obj_w   = (const float*)d_in[7];
  const float* obj_b   = (const float*)d_in[8];
  const float* delta_w = (const float*)d_in[9];
  const float* delta_b = (const float*)d_in[10];
  float* out = (float*)d_out;

  const int   HS[5]  = {200, 100, 50, 25, 13};
  const float STRD[5] = {4.f, 8.f, 16.f, 32.f, 64.f};
  const float SZ[5]  = {32.f, 64.f, 128.f, 256.f, 512.f};
  const int   LOFF[5] = {0, 120000, 150000, 157500, 159375};
  int HWs[5], Mps[5];
  for (int l = 0; l < 5; ++l) { HWs[l] = HS[l] * HS[l]; Mps[l] = ((HWs[l] + 15) / 16) * 16; }

  char* ws = (char*)d_ws;
  size_t off = 0;
  auto walloc = [&](size_t bytes) { size_t o = off; off = (off + bytes + 255) & ~(size_t)255; return o; };
  size_t o_wb2 = walloc((size_t)72 * 256 * 32 * 2);
  size_t o_wh  = walloc((size_t)8 * 16 * 32 * 2);
  size_t o_pad[5], o_hout[5];
  for (int l = 0; l < 5; ++l) o_pad[l]  = walloc((size_t)NB * (HS[l] + 2) * (HS[l] + 2) * CCH * 2);
  for (int l = 0; l < 5; ++l) o_hout[l] = walloc((size_t)NB * Mps[l] * 16 * 4);
  size_t o_sc = walloc((size_t)NB * TA * 4);
  size_t o_bx = walloc((size_t)NB * TA * 16);
  size_t o_cs = walloc((size_t)NB * MCAND * 4);
  size_t o_cb = walloc((size_t)NB * MCAND * 16);
  size_t o_cl = walloc((size_t)NB * MCAND * 4);
  size_t o_ss = walloc((size_t)NB * MCAND * 4);
  size_t o_sb = walloc((size_t)NB * MCAND * 16);
  size_t o_sl = walloc((size_t)NB * MCAND * 4);

  pack_w_kernel<<<(72 * 256 * 32 + 255) / 256, 256, 0, stream>>>(conv_w, (bf16*)(ws + o_wb2));
  pack_head_kernel<<<16, 256, 0, stream>>>(obj_w, delta_w, (bf16*)(ws + o_wh));

  for (int l = 0; l < 5; ++l) {
    long tp = (long)NB * (HS[l] + 2) * (HS[l] + 2) * CCH;
    pad_kernel<<<(int)((tp + 255) / 256), 256, 0, stream>>>(f[l], (bf16*)(ws + o_pad[l]), HS[l], HS[l]);
  }
  for (int l = 0; l < 5; ++l) {
    conv_head_kernel<<<dim3(Mps[l] / 16, NB), 256, 0, stream>>>(
        (const bf16*)(ws + o_pad[l]), (const bf16*)(ws + o_wb2), (const bf16*)(ws + o_wh),
        conv_b, obj_b, delta_b, (float*)(ws + o_hout[l]), HS[l], HS[l], HWs[l], Mps[l]);
  }
  for (int l = 0; l < 5; ++l) {
    int tot = NB * HWs[l] * NA;
    decode_kernel<<<(tot + 255) / 256, 256, 0, stream>>>(
        (const float*)(ws + o_hout[l]), (float*)(ws + o_sc), (float*)(ws + o_bx),
        HS[l], HS[l], HWs[l], Mps[l], STRD[l], SZ[l], LOFF[l]);
  }
  topk_kernel<<<dim3(5, NB), 256, 0, stream>>>(
      (float*)(ws + o_sc), (const float*)(ws + o_bx),
      (float*)(ws + o_cs), (float*)(ws + o_cb), (int*)(ws + o_cl));
  sort_kernel<<<NB, 1024, 0, stream>>>(
      (const float*)(ws + o_cs), (const float*)(ws + o_cb), (const int*)(ws + o_cl),
      (float*)(ws + o_ss), (float*)(ws + o_sb), (int*)(ws + o_sl));
  nms_kernel<<<NB, 1024, 0, stream>>>(
      (const float*)(ws + o_ss), (const float*)(ws + o_sb), (const int*)(ws + o_sl), out);
}
